// MixtureOfExperts_19353122636013
// MI455X (gfx1250) — compile-verified, matrix operand order checked by reading, not run on hardware
//
#include <hip/hip_runtime.h>
#include <hip/hip_bf16.h>
#include <math.h>

// MI455X / gfx1250 Mixture-of-Experts (top-2 routed): bf16 WMMA + TDM async loads.

typedef __bf16 bf16_t;
typedef __attribute__((ext_vector_type(16))) __bf16 v16bf;
typedef __attribute__((ext_vector_type(8)))  __bf16 v8bf;
typedef __attribute__((ext_vector_type(8)))  float  v8f;
typedef __attribute__((ext_vector_type(4)))  unsigned int u32x4;
typedef __attribute__((ext_vector_type(8)))  int          i32x8;
typedef __attribute__((ext_vector_type(4)))  int          i32x4;

#define N_TOK 8192
#define DIM   1024
#define HID   4096
#define NEXP  8
#define TOPK  2
#define TILE_M 128
#define TILE_N 128
#define TILE_K 32
#define LDS_STRIDE 40   // bf16 elems per row: 32 data + 8 pad -> 80B stride (TDM pad matches)
#define MAX_TILES 64

// ---- workspace layout (byte offsets) ----
static const size_t OFF_COUNTS  = 0;                          // 8  int
static const size_t OFF_CURSORS = 64;                         // 8  int
static const size_t OFF_OFFSETS = 128;                        // 9  int
static const size_t OFF_TOPIDX  = 256;                        // 2N int
static const size_t OFF_SLOTTOK = OFF_TOPIDX  + (size_t)2*N_TOK*4;
static const size_t OFF_TOPW    = OFF_SLOTTOK + (size_t)2*N_TOK*4;
static const size_t OFF_SLOTW   = OFF_TOPW    + (size_t)2*N_TOK*4;
static const size_t OFF_HBUF    = ((OFF_SLOTW + (size_t)2*N_TOK*4 + 255) / 256) * 256;
// hbuf: (2N + TILE_M slack) x HID bf16 ~= 128 MB

// ---- LDS byte offset of a __shared__ pointer (ptrtoint of addrspace(3)) ----
__device__ __forceinline__ unsigned lds_off(const void* p) {
    return (unsigned)(uintptr_t)(__attribute__((address_space(3))) const char*)p;
}

// ---- TDM: async 2D bf16 tile (rows x 32 elems) global -> LDS, padded to 80B rows ----
__device__ __forceinline__ void tdm_load_tile_bf16(unsigned lds_addr, const void* gptr,
                                                   unsigned rows, unsigned row_stride_elems) {
    unsigned long long ga = (unsigned long long)(uintptr_t)gptr;
    u32x4 g0;
    g0[0] = 1u;                                          // count=1 (valid), no gather
    g0[1] = lds_addr;                                    // lds_addr
    g0[2] = (unsigned)(ga & 0xffffffffull);              // global_addr lo
    g0[3] = (unsigned)((ga >> 32) & 0x01ffffffull)       // global_addr hi (57-bit)
          | (2u << 30);                                  // type = 2 (image)
    i32x8 g1;
    g1[0] = (int)((1u << 16)        // data_size = 1 -> 2 bytes
                | (1u << 20)        // pad_enable
                | (3u << 22)        // pad_interval: 16 DWORDs (64B)
                | (3u << 25));      // pad_amount:   4 DWORDs (16B) -> 80B row stride
    g1[1] = (int)((65536u & 0xffffu) << 16);             // tensor_dim0 lo16 (dim0=65536)
    g1[2] = (int)((65536u >> 16) | ((65536u & 0xffffu) << 16)); // dim0 hi / dim1 lo
    g1[3] = (int)((65536u >> 16) | (32u << 16));         // dim1 hi | tile_dim0 = 32
    g1[4] = (int)(rows & 0xffffu);                       // tile_dim1 = rows, tile_dim2 = 0
    g1[5] = (int)row_stride_elems;                       // tensor_dim0_stride lo32
    g1[6] = 0;
    g1[7] = 0;
    i32x4 gz4 = {};
#if defined(__clang_major__) && (__clang_major__ >= 23)
    i32x8 gz8 = {};
    __builtin_amdgcn_tensor_load_to_lds(g0, g1, gz4, gz4, gz8, 0);
#else
    __builtin_amdgcn_tensor_load_to_lds(g0, g1, gz4, gz4, 0);
#endif
}

// ---------------- zero output + routing counters ----------------
__global__ void moe_zero(float* __restrict__ out, int* __restrict__ ctrl) {
    size_t i = (size_t)blockIdx.x * blockDim.x + threadIdx.x;   // N*D/4 threads
    float4 z; z.x = 0.f; z.y = 0.f; z.z = 0.f; z.w = 0.f;
    ((float4*)out)[i] = z;
    if (i < 48) ctrl[i] = 0;
}

// ---------------- gate: logits, top-2 softmax, counts ----------------
__global__ __launch_bounds__(256) void moe_gate(
    const float* __restrict__ x, const float* __restrict__ Wg,
    int* __restrict__ counts, int* __restrict__ topIdx, float* __restrict__ topW) {
    int wave  = threadIdx.x >> 5;
    int lane  = threadIdx.x & 31;
    int token = blockIdx.x * 8 + wave;

    const float4* x4 = (const float4*)(x + (size_t)token * DIM);
    const float4* g4 = (const float4*)Wg;

    float acc[NEXP];
#pragma unroll
    for (int e = 0; e < NEXP; ++e) acc[e] = 0.f;

    for (int i = lane; i < DIM / 4; i += 32) {
        float4 xv = x4[i];
#pragma unroll
        for (int e = 0; e < NEXP; ++e) {
            float4 wv = g4[e * (DIM / 4) + i];
            acc[e] += xv.x * wv.x + xv.y * wv.y + xv.z * wv.z + xv.w * wv.w;
        }
    }
#pragma unroll
    for (int e = 0; e < NEXP; ++e)
        for (int off = 16; off; off >>= 1)
            acc[e] += __shfl_xor(acc[e], off, 32);

    if (lane == 0) {
        int i0 = 0; float l0 = acc[0];
        for (int e = 1; e < NEXP; ++e) if (acc[e] > l0) { l0 = acc[e]; i0 = e; }
        int i1 = (i0 == 0) ? 1 : 0; float l1 = acc[i1];
        for (int e = 0; e < NEXP; ++e) if (e != i0 && acc[e] > l1) { l1 = acc[e]; i1 = e; }
        float e1  = expf(l1 - l0);      // l0 >= l1
        float inv = 1.f / (1.f + e1);
        topIdx[token * 2 + 0] = i0;  topIdx[token * 2 + 1] = i1;
        topW [token * 2 + 0] = inv;  topW [token * 2 + 1] = e1 * inv;
        atomicAdd(&counts[i0], 1);
        atomicAdd(&counts[i1], 1);
    }
}

// ---------------- exclusive scan over 8 experts ----------------
__global__ void moe_scan(const int* __restrict__ counts, int* __restrict__ offsets) {
    if (threadIdx.x == 0) {
        int s = 0;
        for (int e = 0; e < NEXP; ++e) { offsets[e] = s; s += counts[e]; }
        offsets[NEXP] = s;
    }
}

// ---------------- slot assignment ----------------
__global__ void moe_assign(
    const int* __restrict__ topIdx, const float* __restrict__ topW,
    const int* __restrict__ offsets, int* __restrict__ cursors,
    int* __restrict__ slotTok, float* __restrict__ slotW) {
    int n = blockIdx.x * blockDim.x + threadIdx.x;
    if (n >= N_TOK) return;
#pragma unroll
    for (int k = 0; k < TOPK; ++k) {
        int e   = topIdx[n * 2 + k];
        int pos = atomicAdd(&cursors[e], 1);
        int s   = offsets[e] + pos;
        slotTok[s] = n;
        slotW [s]  = topW[n * 2 + k];
    }
}

// ---------------- GEMM1: h = gelu(X_e @ W1[e]^T + b1[e]) -> bf16 ----------------
__global__ __launch_bounds__(256) void moe_ffn1(
    const float* __restrict__ x, const float* __restrict__ W1, const float* __restrict__ b1,
    const int* __restrict__ counts, const int* __restrict__ offsets,
    const int* __restrict__ slotTok, bf16_t* __restrict__ hbuf) {
    int e = blockIdx.z, tile = blockIdx.y, hT = blockIdx.x;
    int cnt = counts[e];
    if (tile * TILE_M >= cnt) return;
    int base = offsets[e] + tile * TILE_M;

    __shared__ bf16_t As[TILE_M * LDS_STRIDE];
    __shared__ bf16_t Bs[TILE_N * LDS_STRIDE];
    __shared__ int    tok[TILE_M];

    int tid = threadIdx.x;
    if (tid < TILE_M) {
        int r = tile * TILE_M + tid;
        tok[tid] = (r < cnt) ? slotTok[base + tid] : 0;
    }
    __syncthreads();

    int lrow = tid >> 1;           // 0..127: row of A / B^T tile this thread loads
    int half = tid & 1;            // which 16-element half of the 32-wide K strip
    const float* wrow = W1 + ((size_t)e * HID + (size_t)hT * TILE_N + lrow) * DIM + half * 16;
    const float* xrow = x + (size_t)tok[lrow] * DIM + half * 16;
    bf16_t* adst = As + lrow * LDS_STRIDE + half * 16;
    bf16_t* bdst = Bs + lrow * LDS_STRIDE + half * 16;

    int wave = tid >> 5, lane = tid & 31;
    int m  = lane & 15;
    int kh = lane >> 4;
    v8f acc[8] = {};

    // software pipeline: stage K-chunk 0
    float4 nxa[4], nwb[4];
#pragma unroll
    for (int q = 0; q < 4; ++q) {
        nxa[q] = ((const float4*)xrow)[q];
        nwb[q] = ((const float4*)wrow)[q];
    }

    for (int k = 0; k < DIM; k += TILE_K) {
        // convert staged regs -> bf16 LDS tiles
        v8bf pa[2], pb[2];
#pragma unroll
        for (int q = 0; q < 4; ++q) {
            float4 v = nxa[q], w = nwb[q];
            int h = q >> 1, j = (q & 1) * 4;
            pa[h][j + 0] = (bf16_t)v.x; pa[h][j + 1] = (bf16_t)v.y;
            pa[h][j + 2] = (bf16_t)v.z; pa[h][j + 3] = (bf16_t)v.w;
            pb[h][j + 0] = (bf16_t)w.x; pb[h][j + 1] = (bf16_t)w.y;
            pb[h][j + 2] = (bf16_t)w.z; pb[h][j + 3] = (bf16_t)w.w;
        }
        *(v8bf*)(adst) = pa[0]; *(v8bf*)(adst + 8) = pa[1];
        *(v8bf*)(bdst) = pb[0]; *(v8bf*)(bdst + 8) = pb[1];

        // issue next K-chunk's global loads; they complete during the WMMA phase
        if (k + TILE_K < DIM) {
            __builtin_prefetch(wrow + k + 2 * TILE_K, 0, 1);
#pragma unroll
            for (int q = 0; q < 4; ++q) {
                nxa[q] = ((const float4*)(xrow + k + TILE_K))[q];
                nwb[q] = ((const float4*)(wrow + k + TILE_K))[q];
            }
        }
        __syncthreads();

        // preload ALL fragments, then back-to-back WMMAs (one dscnt wait per K-step)
        union F { v16bf v; v8bf h[2]; };
        F af, bf[8];
        const bf16_t* arow = As + (wave * 16 + m) * LDS_STRIDE;
        af.h[0] = *(const v8bf*)(arow + 8 * kh);
        af.h[1] = *(const v8bf*)(arow + 16 + 8 * kh);
#pragma unroll
        for (int t = 0; t < 8; ++t) {
            const bf16_t* brow = Bs + (t * 16 + m) * LDS_STRIDE;
            bf[t].h[0] = *(const v8bf*)(brow + 8 * kh);
            bf[t].h[1] = *(const v8bf*)(brow + 16 + 8 * kh);
        }
#pragma unroll
        for (int t = 0; t < 8; ++t)
            acc[t] = __builtin_amdgcn_wmma_f32_16x16x32_bf16(
                false, af.v, false, bf[t].v, (short)0, acc[t], false, false);
        __syncthreads();
    }

    // epilogue: bias + exact GELU, store bf16 h rows
#pragma unroll
    for (int t = 0; t < 8; ++t) {
        int col = hT * TILE_N + t * 16 + m;
        float bias = b1[e * HID + col];
#pragma unroll
        for (int r = 0; r < 8; ++r) {
            int trow = wave * 16 + r + 8 * kh;     // C layout: VGPR r -> M=r / M=8+r
            if (tile * TILE_M + trow < cnt) {
                float v = acc[t][r] + bias;
                float g = 0.5f * v * (1.0f + erff(v * 0.70710678118654752f));
                hbuf[(size_t)(base + trow) * HID + col] = (bf16_t)g;
            }
        }
    }
}

// ---------------- GEMM2: out += w * (h @ W2[e]^T + b2[e]) ----------------
// A operand (bf16 h rows) streamed by the Tensor Data Mover, double-buffered on TENSORcnt.
__global__ __launch_bounds__(256) void moe_ffn2(
    const bf16_t* __restrict__ hbuf, const float* __restrict__ W2, const float* __restrict__ b2,
    const int* __restrict__ counts, const int* __restrict__ offsets,
    const int* __restrict__ slotTok, const float* __restrict__ slotW,
    float* __restrict__ out) {
    int e = blockIdx.z, tile = blockIdx.y, dT = blockIdx.x;
    int cnt = counts[e];
    if (tile * TILE_M >= cnt) return;
    int base = offsets[e] + tile * TILE_M;

    __shared__ bf16_t Abuf[2][TILE_M * LDS_STRIDE];
    __shared__ bf16_t Bs[TILE_N * LDS_STRIDE];
    __shared__ int    tok[TILE_M];
    __shared__ float  gw [TILE_M];

    int tid = threadIdx.x;
    if (tid < TILE_M) {
        int r = tile * TILE_M + tid;
        tok[tid] = (r < cnt) ? slotTok[base + tid] : 0;
        gw [tid] = (r < cnt) ? slotW [base + tid] : 0.f;
    }

    int lrow = tid >> 1;
    int half = tid & 1;
    const float* wrow = W2 + ((size_t)e * DIM + (size_t)dT * TILE_N + lrow) * HID + half * 16;
    bf16_t* bdst = Bs + lrow * LDS_STRIDE + half * 16;
    const bf16_t* hTile = hbuf + (size_t)base * HID;   // rows may run past cnt: ws has slack

    int wave = tid >> 5, lane = tid & 31;
    int m  = lane & 15;
    int kh = lane >> 4;
    v8f acc[8] = {};

    // TDM prologue: kick tile 0 into Abuf[0]
    if (wave == 0)
        tdm_load_tile_bf16(lds_off(&Abuf[0][0]), hTile, TILE_M, HID);

    // stage B K-chunk 0
    float4 nwb[4];
#pragma unroll
    for (int q = 0; q < 4; ++q) nwb[q] = ((const float4*)wrow)[q];

    const int nIter = HID / TILE_K;   // 128
    for (int i = 0; i < nIter; ++i) {
        int k   = i * TILE_K;
        int buf = i & 1;

        // async: DMA tile i+1 while we convert/compute tile i
        if (wave == 0 && (i + 1) < nIter)
            tdm_load_tile_bf16(lds_off(&Abuf[buf ^ 1][0]), hTile + (k + TILE_K), TILE_M, HID);

        v8bf pb[2];
#pragma unroll
        for (int q = 0; q < 4; ++q) {
            float4 w = nwb[q];
            int h = q >> 1, j = (q & 1) * 4;
            pb[h][j + 0] = (bf16_t)w.x; pb[h][j + 1] = (bf16_t)w.y;
            pb[h][j + 2] = (bf16_t)w.z; pb[h][j + 3] = (bf16_t)w.w;
        }
        *(v8bf*)(bdst) = pb[0]; *(v8bf*)(bdst + 8) = pb[1];

        if ((i + 1) < nIter) {
            __builtin_prefetch(wrow + k + 2 * TILE_K, 0, 1);
#pragma unroll
            for (int q = 0; q < 4; ++q) nwb[q] = ((const float4*)(wrow + k + TILE_K))[q];
        }

        // wave 0 gates the barrier on tile i's DMA (allow tile i+1 in flight).
        // SIMM16 must be an immediate, so branch to constant calls.
        if (wave == 0) {
            if ((i + 1) < nIter) __builtin_amdgcn_s_wait_tensorcnt(1);
            else                 __builtin_amdgcn_s_wait_tensorcnt(0);
        }
        __syncthreads();

        union F { v16bf v; v8bf h[2]; };
        F af, bf[8];
        const bf16_t* arow = &Abuf[buf][(wave * 16 + m) * LDS_STRIDE];
        af.h[0] = *(const v8bf*)(arow + 8 * kh);
        af.h[1] = *(const v8bf*)(arow + 16 + 8 * kh);
#pragma unroll
        for (int t = 0; t < 8; ++t) {
            const bf16_t* brow = Bs + (t * 16 + m) * LDS_STRIDE;
            bf[t].h[0] = *(const v8bf*)(brow + 8 * kh);
            bf[t].h[1] = *(const v8bf*)(brow + 16 + 8 * kh);
        }
#pragma unroll
        for (int t = 0; t < 8; ++t)
            acc[t] = __builtin_amdgcn_wmma_f32_16x16x32_bf16(
                false, af.v, false, bf[t].v, (short)0, acc[t], false, false);
        __syncthreads();
    }

    // epilogue: bias + gate weight; exactly two atomic contributions per out element
#pragma unroll
    for (int t = 0; t < 8; ++t) {
        int col = dT * TILE_N + t * 16 + m;
        float bias = b2[e * DIM + col];
#pragma unroll
        for (int r = 0; r < 8; ++r) {
            int trow = wave * 16 + r + 8 * kh;
            if (tile * TILE_M + trow < cnt) {
                float contrib = gw[trow] * (acc[t][r] + bias);
                atomicAdd(&out[(size_t)tok[trow] * DIM + col], contrib);
            }
        }
    }
}

// ---------------- launcher ----------------
extern "C" void kernel_launch(void* const* d_in, const int* in_sizes, int n_in,
                              void* d_out, int out_size, void* d_ws, size_t ws_size,
                              hipStream_t stream) {
    (void)in_sizes; (void)n_in; (void)out_size; (void)ws_size;
    const float* x  = (const float*)d_in[0];
    const float* Wg = (const float*)d_in[1];
    const float* W1 = (const float*)d_in[2];
    const float* b1 = (const float*)d_in[3];
    const float* W2 = (const float*)d_in[4];
    const float* b2 = (const float*)d_in[5];
    float* out = (float*)d_out;

    char* ws = (char*)d_ws;
    int*    counts  = (int*)   (ws + OFF_COUNTS);
    int*    cursors = (int*)   (ws + OFF_CURSORS);
    int*    offsets = (int*)   (ws + OFF_OFFSETS);
    int*    topIdx  = (int*)   (ws + OFF_TOPIDX);
    int*    slotTok = (int*)   (ws + OFF_SLOTTOK);
    float*  topW    = (float*) (ws + OFF_TOPW);
    float*  slotW   = (float*) (ws + OFF_SLOTW);
    bf16_t* hbuf    = (bf16_t*)(ws + OFF_HBUF);

    moe_zero<<<(N_TOK * DIM / 4) / 256, 256, 0, stream>>>(out, counts);
    moe_gate<<<N_TOK / 8, 256, 0, stream>>>(x, Wg, counts, topIdx, topW);
    moe_scan<<<1, 32, 0, stream>>>(counts, offsets);
    moe_assign<<<N_TOK / 256, 256, 0, stream>>>(topIdx, topW, offsets, cursors, slotTok, slotW);
    dim3 g1(HID / TILE_N, MAX_TILES, NEXP);
    moe_ffn1<<<g1, 256, 0, stream>>>(x, W1, b1, counts, offsets, slotTok, hbuf);
    dim3 g2(DIM / TILE_N, MAX_TILES, NEXP);
    moe_ffn2<<<g2, 256, 0, stream>>>(hbuf, W2, b2, counts, offsets, slotTok, slotW, out);
}
